// Attention_3719441678340
// MI455X (gfx1250) — matrix operand order repaired and checked by reading, not compiled, not run on hardware
//
#include <hip/hip_runtime.h>
#include <hip/hip_bf16.h>

#define NHEADS 16
#define HD 128
#define SEQ 2048
#define HID 2048
#define BATCH 2
#define KSTEP 64
#define LPAD 72   // row pitch in halfwords: 144B, 16B-aligned, bank-skewed

typedef _Float16 half8 __attribute__((ext_vector_type(8)));
typedef _Float16 v16h  __attribute__((ext_vector_type(16)));
typedef float    v8f   __attribute__((ext_vector_type(8)));
typedef int      vi4   __attribute__((vector_size(16)));   // matches builtin param type

__device__ __forceinline__ v8f zero8() {
  v8f z = {0.f,0.f,0.f,0.f,0.f,0.f,0.f,0.f};
  return z;
}

// 16-halfword fragment from two 16-byte chunks (ds_read_b128 / global_load_b128 pairs)
__device__ __forceinline__ v16h make_frag(const _Float16* p0, const _Float16* p1) {
  half8 lo = *(const half8*)p0;
  half8 hi = *(const half8*)p1;
  return __builtin_shufflevector(lo, hi, 0,1,2,3,4,5,6,7,8,9,10,11,12,13,14,15);
}

// ---- CDNA5 async global->LDS path (ASYNCcnt), with sync fallback ----
#if __has_builtin(__builtin_amdgcn_global_load_async_to_lds_b128)
#define HAVE_ASYNC 1
typedef __attribute__((address_space(1))) vi4* gvi4_p;
typedef __attribute__((address_space(3))) vi4* lvi4_p;
__device__ __forceinline__ void cp16(const _Float16* g, _Float16* l) {
  __builtin_amdgcn_global_load_async_to_lds_b128(
      (gvi4_p)(uintptr_t)g, (lvi4_p)(uintptr_t)l, 0, 0);
}
__device__ __forceinline__ void wait_async_all() {
#if __has_builtin(__builtin_amdgcn_s_wait_asynccnt)
  __builtin_amdgcn_s_wait_asynccnt(0);
#else
  asm volatile("s_wait_asynccnt 0" ::: "memory");
#endif
}
#else
__device__ __forceinline__ void cp16(const _Float16* g, _Float16* l) {
  __builtin_prefetch(g, 0, 0);
  *(half8*)l = *(const half8*)g;
}
__device__ __forceinline__ void wait_async_all() {}
#endif

__global__ void cvt_f32_to_f16(const float* __restrict__ s, _Float16* __restrict__ d, int n) {
  int i = blockIdx.x * blockDim.x + threadIdx.x;
  int stride = gridDim.x * blockDim.x;
  for (; i < n; i += stride) d[i] = (_Float16)s[i];
}

// C[m,n] = sum_k A[m,k] * W[n,k] + bias[n]   (M=4096, N=K=2048)
// HEADS_OUT: f16 store reshaped to [B][NHEADS][S][HD]; else f32 store [M][N].
template<bool HEADS_OUT>
__global__ __launch_bounds__(256) void gemm128(const _Float16* __restrict__ A,
    const _Float16* __restrict__ W, const float* __restrict__ bias,
    _Float16* __restrict__ out16, float* __restrict__ outf) {
  const int K = HID, N = HID;
  __shared__ _Float16 As[2][128][LPAD];
  __shared__ _Float16 Bs[2][128][LPAD];
  const int tid  = threadIdx.x;
  const int lane = tid & 31, wave = tid >> 5;
  const int wm = wave & 3, wn = wave >> 2;        // 4x2 wave grid -> 32x64 per wave
  const int m0 = blockIdx.y * 128, n0 = blockIdx.x * 128;
  const int lrow = lane & 15, lhi = lane >> 4;

  v8f acc[2][4];
#pragma unroll
  for (int i = 0; i < 2; ++i)
#pragma unroll
    for (int j = 0; j < 4; ++j) acc[i][j] = zero8();

  auto stage = [&](int buf, int k0) {
#pragma unroll
    for (int i = 0; i < 4; ++i) {
      int c = tid + i * 256;            // 1024 chunks of 8 hw = 128x64 tile
      int r = c >> 3, cc = (c & 7) * 8;
      cp16(A + (size_t)(m0 + r) * K + k0 + cc, &As[buf][r][cc]);
      cp16(W + (size_t)(n0 + r) * K + k0 + cc, &Bs[buf][r][cc]);
    }
  };

  stage(0, 0);
  wait_async_all();
  __syncthreads();

  int buf = 0;
  for (int k0 = 0; k0 < K; k0 += KSTEP) {
    if (k0 + KSTEP < K) stage(buf ^ 1, k0 + KSTEP);

    v16h af[2][2], bfr[4][2];
#pragma unroll
    for (int mi = 0; mi < 2; ++mi)
#pragma unroll
      for (int kc = 0; kc < 2; ++kc) {
        const _Float16* p = &As[buf][wm * 32 + mi * 16 + lrow][kc * 32 + lhi * 8];
        af[mi][kc] = make_frag(p, p + 16);          // A: K 0-7/8-15 then +16
      }
#pragma unroll
    for (int ni = 0; ni < 4; ++ni)
#pragma unroll
      for (int kc = 0; kc < 2; ++kc) {
        const _Float16* p = &Bs[buf][wn * 64 + ni * 16 + lrow][kc * 32 + lhi * 8];
        bfr[ni][kc] = make_frag(p, p + 16);         // B: K 0-7/8-15 then +16
      }
#pragma unroll
    for (int kc = 0; kc < 2; ++kc)
#pragma unroll
      for (int mi = 0; mi < 2; ++mi)
#pragma unroll
        for (int ni = 0; ni < 4; ++ni)
          acc[mi][ni] = __builtin_amdgcn_wmma_f32_16x16x32_f16(
              false, af[mi][kc], false, bfr[ni][kc], (short)0, acc[mi][ni], false, false);

    wait_async_all();
    __syncthreads();
    buf ^= 1;
  }

#pragma unroll
  for (int mi = 0; mi < 2; ++mi)
#pragma unroll
    for (int ni = 0; ni < 4; ++ni) {
      int col = n0 + wn * 64 + ni * 16 + lrow;
      float bv = bias[col];
#pragma unroll
      for (int j = 0; j < 8; ++j) {
        int row = m0 + wm * 32 + mi * 16 + j + lhi * 8;
        float v = acc[mi][ni][j] + bv;
        if (HEADS_OUT) {
          int b = row >> 11, s = row & (SEQ - 1);
          int hh = col >> 7, dd = col & (HD - 1);
          out16[(((size_t)(b * NHEADS + hh)) * SEQ + s) * HD + dd] = (_Float16)v;
        } else {
          outf[(size_t)row * N + col] = v;
        }
      }
    }
}

// Flash attention forward: Q,K,V f16 [B][NHEADS][S][HD] -> O f16 [B][S][HID]
__global__ __launch_bounds__(256) void attn_fwd(const _Float16* __restrict__ Qp,
    const _Float16* __restrict__ Kp, const _Float16* __restrict__ Vp,
    _Float16* __restrict__ Op) {
  __shared__ _Float16 Ks[2][64 * HD];    // 64 keys x 128, row-major (async)
  __shared__ _Float16 Vt[2][HD * 64];    // V transposed [d][k] (sync scatter)
  __shared__ _Float16 Ps[8][16 * 64];    // per-wave P transpose buffer
  const int tid = threadIdx.x, lane = tid & 31, wave = tid >> 5;
  const int lrow = lane & 15, lhi = lane >> 4;
  const int qb = blockIdx.x, bh = blockIdx.y;
  const int bidx = bh >> 4, h = bh & 15;
  const size_t base = (size_t)bh * SEQ * HD;
  const int q0 = qb * 128 + wave * 16;
  const float SCALE2 = 0.08838834764831845f * 1.4426950408889634f; // d^-0.5 * log2(e)

  // Resident Q fragments (A-layout, loaded straight from global)
  v16h qf[4];
#pragma unroll
  for (int dc = 0; dc < 4; ++dc) {
    const _Float16* p = Qp + base + (size_t)(q0 + lrow) * HD + dc * 32 + lhi * 8;
    qf[dc] = make_frag(p, p + 16);
  }

  v8f oa[8];
#pragma unroll
  for (int n = 0; n < 8; ++n) oa[n] = zero8();
  float mrow[8], lsum[8];
#pragma unroll
  for (int j = 0; j < 8; ++j) { mrow[j] = -1e30f; lsum[j] = 0.f; }

  auto stage = [&](int buf, int k0) {
#pragma unroll
    for (int i = 0; i < 4; ++i) {
      int c = tid + i * 256;            // 1024 chunks of 8 hw = 64x128 tile
      int r = c >> 4, cc = (c & 15) * 8;
      cp16(Kp + base + (size_t)(k0 + r) * HD + cc, &Ks[buf][r * HD + cc]);
      half8 v = *(const half8*)(Vp + base + (size_t)(k0 + r) * HD + cc);
#pragma unroll
      for (int e = 0; e < 8; ++e) Vt[buf][(cc + e) * 64 + r] = v[e];
    }
  };

  const int nkt = (qb + 1) * 2;          // causal: keys up to end of this q-block
  stage(0, 0);
  wait_async_all();
  __syncthreads();

  int buf = 0;
  for (int kt = 0; kt < nkt; ++kt) {
    const int k0 = kt * 64;
    if (kt + 1 < nkt) stage(buf ^ 1, (kt + 1) * 64);

    // Scores: S = Q K^T for this wave's 16 q-rows x 64 keys
    v8f sf[4];
#pragma unroll
    for (int f = 0; f < 4; ++f) {
      sf[f] = zero8();
#pragma unroll
      for (int dc = 0; dc < 4; ++dc) {
        const _Float16* p = &Ks[buf][(f * 16 + lrow) * HD + dc * 32 + lhi * 8];
        v16h bf = make_frag(p, p + 16);
        sf[f] = __builtin_amdgcn_wmma_f32_16x16x32_f16(
            false, qf[dc], false, bf, (short)0, sf[f], false, false);
      }
    }

    // Scale (log2 domain), causal mask, per-row max
    float smax[8];
#pragma unroll
    for (int j = 0; j < 8; ++j) smax[j] = -1e30f;
#pragma unroll
    for (int f = 0; f < 4; ++f) {
      int kcol = k0 + f * 16 + lrow;
#pragma unroll
      for (int j = 0; j < 8; ++j) {
        int qrow = q0 + j + lhi * 8;
        float v = sf[f][j] * SCALE2;
        if (kcol > qrow) v = -1e30f;
        sf[f][j] = v;
        smax[j] = fmaxf(smax[j], v);
      }
    }
    // Online softmax update (row lives across 16 lanes)
#pragma unroll
    for (int j = 0; j < 8; ++j) {
#pragma unroll
      for (int off = 1; off < 16; off <<= 1)
        smax[j] = fmaxf(smax[j], __shfl_xor(smax[j], off, 16));
      float mnew = fmaxf(mrow[j], smax[j]);
      float corr = exp2f(mrow[j] - mnew);
      mrow[j] = mnew;
      lsum[j] *= corr;
#pragma unroll
      for (int n = 0; n < 8; ++n) oa[n][j] *= corr;
    }
    // P = exp2(S - m); accumulate partial row sums; stash P to per-wave LDS
#pragma unroll
    for (int f = 0; f < 4; ++f) {
#pragma unroll
      for (int j = 0; j < 8; ++j) {
        float pv = exp2f(sf[f][j] - mrow[j]);
        lsum[j] += pv;
        Ps[wave][(j + lhi * 8) * 64 + f * 16 + lrow] = (_Float16)pv;
      }
    }
    asm volatile("s_wait_dscnt 0" ::: "memory");  // intra-wave DS RAW fence

    // Reload P as A-fragments (LDS transpose) and do O += P V
    v16h pa[2];
#pragma unroll
    for (int kc = 0; kc < 2; ++kc) {
      const _Float16* p = &Ps[wave][lrow * 64 + kc * 32 + lhi * 8];
      pa[kc] = make_frag(p, p + 16);
    }
#pragma unroll
    for (int n = 0; n < 8; ++n)
#pragma unroll
      for (int kc = 0; kc < 2; ++kc) {
        const _Float16* p = &Vt[buf][(n * 16 + lrow) * 64 + kc * 32 + lhi * 8];
        v16h bf = make_frag(p, p + 16);
        oa[n] = __builtin_amdgcn_wmma_f32_16x16x32_f16(
            false, pa[kc], false, bf, (short)0, oa[n], false, false);
      }

    wait_async_all();
    __syncthreads();
    buf ^= 1;
  }

  // Normalize and store O in [B][S][HID] f16 layout
  float inv[8];
#pragma unroll
  for (int j = 0; j < 8; ++j) {
#pragma unroll
    for (int off = 1; off < 16; off <<= 1)
      lsum[j] += __shfl_xor(lsum[j], off, 16);
    inv[j] = 1.0f / lsum[j];
  }
#pragma unroll
  for (int n = 0; n < 8; ++n) {
    int col = h * HD + n * 16 + lrow;
#pragma unroll
    for (int j = 0; j < 8; ++j) {
      int row = q0 + j + lhi * 8;
      Op[((size_t)bidx * SEQ + row) * HID + col] = (_Float16)(oa[n][j] * inv[j]);
    }
  }
}

extern "C" void kernel_launch(void* const* d_in, const int* in_sizes, int n_in,
                              void* d_out, int out_size, void* d_ws, size_t ws_size,
                              hipStream_t stream) {
  const float* x  = (const float*)d_in[0];
  const float* wq = (const float*)d_in[1];
  const float* bq = (const float*)d_in[2];
  const float* wk = (const float*)d_in[3];
  const float* bk = (const float*)d_in[4];
  const float* wv = (const float*)d_in[5];
  const float* bv = (const float*)d_in[6];
  const float* wo = (const float*)d_in[7];
  const float* bo = (const float*)d_in[8];
  float* out = (float*)d_out;

  const size_t MN = (size_t)BATCH * SEQ * HID;  // 8.39M elems
  const size_t WN = (size_t)HID * HID;          // 4.19M elems
  _Float16* w = (_Float16*)d_ws;
  _Float16* x16  = w; w += MN;
  _Float16* wq16 = w; w += WN;
  _Float16* wk16 = w; w += WN;
  _Float16* wv16 = w; w += WN;
  _Float16* wo16 = w; w += WN;
  _Float16* Q16  = w; w += MN;
  _Float16* K16  = w; w += MN;
  _Float16* V16  = w; w += MN;
  _Float16* O16  = w; w += MN;

  cvt_f32_to_f16<<<2048, 256, 0, stream>>>(x,  x16,  (int)MN);
  cvt_f32_to_f16<<<1024, 256, 0, stream>>>(wq, wq16, (int)WN);
  cvt_f32_to_f16<<<1024, 256, 0, stream>>>(wk, wk16, (int)WN);
  cvt_f32_to_f16<<<1024, 256, 0, stream>>>(wv, wv16, (int)WN);
  cvt_f32_to_f16<<<1024, 256, 0, stream>>>(wo, wo16, (int)WN);

  dim3 g(HID / 128, (BATCH * SEQ) / 128);  // (16, 32)
  gemm128<true><<<g, 256, 0, stream>>>(x16, wq16, bq, Q16, nullptr);
  gemm128<true><<<g, 256, 0, stream>>>(x16, wk16, bk, K16, nullptr);
  gemm128<true><<<g, 256, 0, stream>>>(x16, wv16, bv, V16, nullptr);

  attn_fwd<<<dim3(SEQ / 128, BATCH * NHEADS), 256, 0, stream>>>(Q16, K16, V16, O16);

  gemm128<false><<<g, 256, 0, stream>>>(O16, wo16, bo, nullptr, out);
}
